// RNNGraphConvModule_19791209300764
// MI455X (gfx1250) — matrix operand order repaired and checked by reading, not compiled, not run on hardware
//
#include <hip/hip_runtime.h>
#include <stdint.h>

// ---------------------------------------------------------------------------
// Problem constants (match reference)
// ---------------------------------------------------------------------------
constexpr int N_NODES = 20000;
constexpr int N_EDGES = 200000;
constexpr int NC      = 32;    // node feature dim
constexpr int FDIM    = 13;    // edge feature dim
constexpr int FHID    = 64;    // filter hidden dim
constexpr int NREP    = 10;
constexpr int OUTC    = NC * (NREP + 1);   // 352

// workspace layout (bytes, 256-aligned)
constexpr size_t OFF_W    = 0;                                   // bf16 [E][32(o)][32(i)]  409.6 MB
constexpr size_t OFF_H1   = OFF_W    + (size_t)N_EDGES * 1024 * 2;   // bf16 [E][64]        25.6 MB
constexpr size_t OFF_FW2B = OFF_H1   + (size_t)N_EDGES * 64 * 2;     // bf16 swizzled B     128 KB
constexpr size_t OFF_AGG  = OFF_FW2B + 65536 * 2;                    // f32  [N][32]
constexpr size_t OFF_HX   = OFF_AGG  + (size_t)N_NODES * NC * 4;     // f32  [N][32]
constexpr size_t OFF_DEG  = OFF_HX   + (size_t)N_NODES * NC * 4;     // i32  [N]
constexpr size_t OFF_INV  = OFF_DEG  + ((size_t)N_NODES * 4 + 255 & ~(size_t)255); // f32 [N]

// ---------------------------------------------------------------------------
// bf16 helpers (manual, no type headaches)
// ---------------------------------------------------------------------------
__device__ __forceinline__ unsigned short f2bf(float f) {
    unsigned u = __float_as_uint(f);
    u += 0x7FFFu + ((u >> 16) & 1u);   // round-to-nearest-even
    return (unsigned short)(u >> 16);
}
__device__ __forceinline__ float bf_lo(unsigned w) { return __uint_as_float(w << 16); }
__device__ __forceinline__ float bf_hi(unsigned w) { return __uint_as_float(w & 0xFFFF0000u); }

// WMMA fragment types (gfx1250)
typedef __bf16 v16bf __attribute__((ext_vector_type(16)));
typedef float  v8f   __attribute__((ext_vector_type(8)));

union Frag { v16bf v; uint4 q[2]; };
union Row  { uint4 q4[4]; unsigned u[16]; };

// ---------------------------------------------------------------------------
// Kernel 1: h1 = relu(edge_feats @ fW1 + fb1)   -> bf16 [E][64]
// one thread per (edge, hidden) element; fW1 (3.3 KB) lives in L0/L2
// ---------------------------------------------------------------------------
__global__ __launch_bounds__(256)
void k_filter1(const float* __restrict__ ef, const float* __restrict__ fW1,
               const float* __restrict__ fb1, unsigned short* __restrict__ h1) {
    int t = blockIdx.x * 256 + threadIdx.x;          // exact grid: E*64 threads
    int e = t >> 6, j = t & 63;
    const float* f = ef + (size_t)e * FDIM;
    float acc = fb1[j];
#pragma unroll
    for (int k = 0; k < FDIM; ++k) acc = fmaf(f[k], fW1[k * FHID + j], acc);
    h1[t] = f2bf(fmaxf(acc, 0.0f));
}

// ---------------------------------------------------------------------------
// Kernel 2: pre-swizzle fW2 (64x1024 f32) into WMMA-B bf16 lane layout:
//   fW2B[((c*64 + ot)*32 + lane)*16 + h] = fW2[(c*32 + (lane<16?0:16) + h)][ot*16 + (lane&15)]
// ---------------------------------------------------------------------------
__global__ __launch_bounds__(256)
void k_prepB(const float* __restrict__ fW2, unsigned short* __restrict__ fW2B) {
    int t = blockIdx.x * 256 + threadIdx.x;          // exact grid: 65536 threads
    int h = t & 15, lane = (t >> 4) & 31, ot = (t >> 9) & 63, c = t >> 15;
    int row = c * 32 + ((lane >> 4) << 4) + h;       // K index (0..63)
    int col = ot * 16 + (lane & 15);                 // N index (0..1023)
    fW2B[t] = f2bf(fW2[(size_t)row * 1024 + col]);
}

// ---------------------------------------------------------------------------
// Kernel 3: node init — copy hx, write out[:,0:32], zero agg, zero deg
// ---------------------------------------------------------------------------
__global__ __launch_bounds__(256)
void k_init_nodes(const float* __restrict__ hx_in, float* __restrict__ hx,
                  float* __restrict__ agg, int* __restrict__ deg,
                  float* __restrict__ out) {
    int t = blockIdx.x * 256 + threadIdx.x;          // exact grid: N*32 threads
    int n = t >> 5, l = t & 31;
    float v = hx_in[t];
    hx[t]  = v;
    agg[t] = 0.0f;
    out[(size_t)n * OUTC + l] = v;
    if (l == 0) deg[n] = 0;
}

__global__ __launch_bounds__(256)
void k_degree(const int* __restrict__ dst, int* __restrict__ deg) {
    int t = blockIdx.x * 256 + threadIdx.x;
    if (t < N_EDGES) atomicAdd(&deg[dst[t]], 1);
}

__global__ __launch_bounds__(256)
void k_invdeg(const int* __restrict__ deg, float* __restrict__ invd) {
    int t = blockIdx.x * 256 + threadIdx.x;
    if (t < N_NODES) invd[t] = 1.0f / (float)max(deg[t], 1);
}

// ---------------------------------------------------------------------------
// Kernel 4: filter GEMM  weights = h1[E,64] @ fW2[64,1024] + fb2  (bf16 WMMA)
// One wave per 16-edge tile; A (16x64 bf16) in registers for the whole
// 64-N-tile loop; B fragments are contiguous per-lane loads from fW2B.
// D stored transposed as w[e][o][i] bf16 for the matvec kernel.
// ---------------------------------------------------------------------------
__global__ __launch_bounds__(128)
void k_gemm(const unsigned short* __restrict__ h1,
            const unsigned short* __restrict__ fW2B,
            const float* __restrict__ fb2,
            unsigned short* __restrict__ w) {
    int lane = threadIdx.x & 31;
    int wid  = (blockIdx.x * 128 + threadIdx.x) >> 5;   // 0..12499 (exact)
    int eb   = wid * 16;
    int m15  = lane & 15;
    int hi16 = lane >> 4;                               // 0 or 1

    // A fragments: 16x32 bf16 per K-chunk.  lane<16: K {0-15}; lane>=16: K {16-31}
    const unsigned short* arow = h1 + (size_t)(eb + m15) * 64;
    int kb = hi16 * 16;
    Frag a0, a1;
    a0.q[0] = *(const uint4*)(arow + kb);
    a0.q[1] = *(const uint4*)(arow + kb + 8);
    a1.q[0] = *(const uint4*)(arow + 32 + kb);
    a1.q[1] = *(const uint4*)(arow + 32 + kb + 8);

    for (int ot = 0; ot < 64; ++ot) {
        Frag b0, b1;
        const unsigned short* bp0 = fW2B + ((size_t)(ot) * 32 + lane) * 16;
        const unsigned short* bp1 = fW2B + ((size_t)(64 + ot) * 32 + lane) * 16;
        b0.q[0] = *(const uint4*)(bp0); b0.q[1] = *(const uint4*)(bp0 + 8);
        b1.q[0] = *(const uint4*)(bp1); b1.q[1] = *(const uint4*)(bp1 + 8);

        v8f d = {0.f, 0.f, 0.f, 0.f, 0.f, 0.f, 0.f, 0.f};
        d = __builtin_amdgcn_wmma_f32_16x16x32_bf16(false, a0.v, false, b0.v,
                                                    (short)0, d, false, false);
        d = __builtin_amdgcn_wmma_f32_16x16x32_bf16(false, a1.v, false, b1.v,
                                                    (short)0, d, false, false);

        int col  = ot * 16 + m15;       // 0..1023 = i*32 + o
        float bias = fb2[col];
        int ii = col >> 5, oo = col & 31;
#pragma unroll
        for (int r = 0; r < 8; ++r) {   // D: lanes0-15 M=r, lanes16-31 M=8+r
            int e = eb + r + (hi16 << 3);
            w[(size_t)e * 1024 + oo * 32 + ii] = f2bf(d[r] + bias);
        }
    }
}

// ---------------------------------------------------------------------------
// Kernel 5: per-edge matvec + scatter.  One wave per edge, lane = output o.
// Lane streams its bf16 weight row (32 elems = 4 x b128), broadcasts the
// gathered source features via ds_bpermute, atomically adds into agg[dst].
// This is the bandwidth-dominant kernel: 409.6 MB of bf16 weights per iter.
// ---------------------------------------------------------------------------
__global__ __launch_bounds__(256)
void k_msg(const unsigned short* __restrict__ w, const float* __restrict__ hx,
           const int* __restrict__ src, const int* __restrict__ dst,
           float* __restrict__ agg) {
    int lane = threadIdx.x & 31;
    int e = (blockIdx.x * 256 + threadIdx.x) >> 5;      // exact: E waves
    int s = src[e];
    int d = dst[e];
    float x = hx[(size_t)s * NC + lane];                // coalesced 128B gather

    Row r;
    const uint4* wr = (const uint4*)(w + (size_t)e * 1024 + (size_t)lane * 32);
    r.q4[0] = wr[0]; r.q4[1] = wr[1]; r.q4[2] = wr[2]; r.q4[3] = wr[3];

    float acc = 0.0f;
#pragma unroll
    for (int p = 0; p < 16; ++p) {
        unsigned wb = r.u[p];                           // bf16 pair (i=2p, i=2p+1)
        float xa = __shfl(x, 2 * p);
        float xb = __shfl(x, 2 * p + 1);
        acc = fmaf(bf_lo(wb), xa, acc);
        acc = fmaf(bf_hi(wb), xb, acc);
    }
    atomicAdd(&agg[(size_t)d * NC + lane], acc);
}

// ---------------------------------------------------------------------------
// Kernel 6: GRU cell per node (wave per node), Wi/Wh staged in LDS.
// Also applies 1/deg normalization and re-zeroes agg for the next iteration.
// ---------------------------------------------------------------------------
__global__ __launch_bounds__(256)
void k_gru(float* __restrict__ agg, const float* __restrict__ invd,
           float* __restrict__ hx,
           const float* __restrict__ Wi, const float* __restrict__ Wh,
           const float* __restrict__ bi, const float* __restrict__ bh,
           float* __restrict__ out, int it) {
    __shared__ float sWi[32 * 96];
    __shared__ float sWh[32 * 96];
    for (int t = threadIdx.x; t < 32 * 96; t += 256) { sWi[t] = Wi[t]; sWh[t] = Wh[t]; }
    __syncthreads();

    int lane = threadIdx.x & 31;
    int n = (blockIdx.x * 256 + threadIdx.x) >> 5;      // exact: N waves

    float a = agg[(size_t)n * NC + lane] * invd[n];
    agg[(size_t)n * NC + lane] = 0.0f;                  // reset for next iter
    float h = hx[(size_t)n * NC + lane];

    float ir = bi[lane], iz = bi[32 + lane], inn = bi[64 + lane];
    float hr = bh[lane], hz = bh[32 + lane], hn  = bh[64 + lane];
#pragma unroll
    for (int i = 0; i < 32; ++i) {
        float ai = __shfl(a, i);
        float hi = __shfl(h, i);
        ir  = fmaf(ai, sWi[i * 96 + lane],      ir);
        iz  = fmaf(ai, sWi[i * 96 + 32 + lane], iz);
        inn = fmaf(ai, sWi[i * 96 + 64 + lane], inn);
        hr  = fmaf(hi, sWh[i * 96 + lane],      hr);
        hz  = fmaf(hi, sWh[i * 96 + 32 + lane], hz);
        hn  = fmaf(hi, sWh[i * 96 + 64 + lane], hn);
    }
    float rg = 1.0f / (1.0f + __expf(-(ir + hr)));
    float zg = 1.0f / (1.0f + __expf(-(iz + hz)));
    float ng = tanhf(inn + rg * hn);
    float nh = (1.0f - zg) * ng + zg * h;

    hx[(size_t)n * NC + lane] = nh;
    out[(size_t)n * OUTC + (size_t)(it + 1) * NC + lane] = nh;
}

// ---------------------------------------------------------------------------
// Launch: inputs in setup_inputs() order.
// ws requirement ~441 MB (reference itself materializes 819 MB of weights).
// ---------------------------------------------------------------------------
extern "C" void kernel_launch(void* const* d_in, const int* in_sizes, int n_in,
                              void* d_out, int out_size, void* d_ws, size_t ws_size,
                              hipStream_t stream) {
    const float* hx_in = (const float*)d_in[0];
    const int*   eidx  = (const int*)d_in[1];
    const float* efeat = (const float*)d_in[2];
    const float* fW1   = (const float*)d_in[3];
    const float* fb1   = (const float*)d_in[4];
    const float* fW2   = (const float*)d_in[5];
    const float* fb2   = (const float*)d_in[6];
    const float* Wi    = (const float*)d_in[7];
    const float* Wh    = (const float*)d_in[8];
    const float* bi    = (const float*)d_in[9];
    const float* bh    = (const float*)d_in[10];
    float* out = (float*)d_out;

    char* ws = (char*)d_ws;
    unsigned short* w    = (unsigned short*)(ws + OFF_W);
    unsigned short* h1   = (unsigned short*)(ws + OFF_H1);
    unsigned short* fW2B = (unsigned short*)(ws + OFF_FW2B);
    float* agg  = (float*)(ws + OFF_AGG);
    float* hx   = (float*)(ws + OFF_HX);
    int*   deg  = (int*)(ws + OFF_DEG);
    float* invd = (float*)(ws + OFF_INV);

    const int* src = eidx;
    const int* dst = eidx + N_EDGES;

    // one-time (per call) prep
    k_filter1<<<(N_EDGES * 64) / 256, 256, 0, stream>>>(efeat, fW1, fb1, h1);
    k_prepB<<<65536 / 256, 256, 0, stream>>>(fW2, fW2B);
    k_init_nodes<<<(N_NODES * NC) / 256, 256, 0, stream>>>(hx_in, hx, agg, deg, out);
    k_degree<<<(N_EDGES + 255) / 256, 256, 0, stream>>>(dst, deg);
    k_invdeg<<<(N_NODES + 255) / 256, 256, 0, stream>>>(deg, invd);
    k_gemm<<<(N_EDGES / 16) / 4, 128, 0, stream>>>(h1, fW2B, fb2, w);

    // 10 propagation + GRU iterations
    for (int it = 0; it < NREP; ++it) {
        k_msg<<<N_EDGES / 8, 256, 0, stream>>>(w, hx, src, dst, agg);
        k_gru<<<N_NODES / 8, 256, 0, stream>>>(agg, invd, hx, Wi, Wh, bi, bh, out, it);
    }
}
